// SCC3D_9964324126880
// MI455X (gfx1250) — matrix operand order repaired and checked by reading, not compiled, not run on hardware
//
#include <hip/hip_runtime.h>
#include <hip/hip_bf16.h>

// ---------------------------------------------------------------------------
// Swin-3D window attention for MI455X (gfx1250, wave32, WMMA bf16 path)
// B_=192 windows, N=256 tokens, C=256, H=8 heads, hd=32, nW=64 masks.
// ---------------------------------------------------------------------------

typedef __bf16 bf16;
typedef __attribute__((ext_vector_type(16))) __bf16 v16bf;
typedef __attribute__((ext_vector_type(8)))  __bf16 v8bf;
typedef __attribute__((ext_vector_type(8)))  float   v8f;

#define NWIN   192
#define NTOK   256
#define CDIM   256
#define HEADS  8
#define HD     32
#define NMASK  64
#define MTOT   (NWIN * NTOK)          // 49152 rows for the big GEMMs

__device__ __forceinline__ bf16 f2bf(float f) {
    unsigned u = __float_as_uint(f);
    unsigned r = (u + 0x7FFFu + ((u >> 16) & 1u)) >> 16;
    unsigned short s = (unsigned short)r;
    return __builtin_bit_cast(__bf16, s);
}

__device__ __forceinline__ v8f vzero8() {
    v8f z;
#pragma unroll
    for (int i = 0; i < 8; ++i) z[i] = 0.0f;
    return z;
}

// A operand fragment (16x32 bf16, lane = row of A, K contiguous in memory).
// ISA layout: elems 0..7 -> K = half*8 + 0..7 ; elems 8..15 -> K = 16 + half*8 + 0..7
__device__ __forceinline__ v16bf load_frag_a(const bf16* __restrict__ rowptr, int half) {
    v8bf lo = *(const v8bf*)(rowptr + half * 8);
    v8bf hi = *(const v8bf*)(rowptr + 16 + half * 8);
    v16bf r;
#pragma unroll
    for (int i = 0; i < 8; ++i) { r[i] = lo[i]; r[8 + i] = hi[i]; }
    return r;
}

// B operand fragment (32x16 bf16, lane = column of B). Source stored as B^T
// rows (K contiguous). ISA layout: elems 0..7 -> K = half*8 + 0..7 ; elems 8..15 -> K = 16 + half*8 + 0..7
__device__ __forceinline__ v16bf load_frag_b(const bf16* __restrict__ colptr, int half) {
    v8bf lo = *(const v8bf*)(colptr + half * 8);
    v8bf hi = *(const v8bf*)(colptr + 16 + half * 8);
    v16bf r;
#pragma unroll
    for (int i = 0; i < 8; ++i) { r[i] = lo[i]; r[8 + i] = hi[i]; }
    return r;
}

__device__ __forceinline__ v8f wmma_bf(v16bf a, v16bf b, v8f c) {
    return __builtin_amdgcn_wmma_f32_16x16x32_bf16(
        /*neg_a=*/false, a, /*neg_b=*/false, b,
        /*c_mod=*/(short)0, c, /*reuse_a=*/false, /*reuse_b=*/false);
}

// ---------------------------------------------------------------------------
// Prep kernels
// ---------------------------------------------------------------------------
__global__ void f32_to_bf16_kernel(const float* __restrict__ in,
                                   bf16* __restrict__ out, int n) {
    int i = blockIdx.x * blockDim.x + threadIdx.x;
    if (i < n) out[i] = f2bf(in[i]);
}

// biasH[h][n][m] = table[idx[n*256+m]*8 + h]
__global__ void bias_gather_kernel(const float* __restrict__ table,
                                   const int* __restrict__ idx,
                                   float* __restrict__ biasH) {
    int i = blockIdx.x * blockDim.x + threadIdx.x;   // 0..65535
    if (i < NTOK * NTOK) {
        int t = idx[i];
#pragma unroll
        for (int h = 0; h < HEADS; ++h)
            biasH[h * (NTOK * NTOK) + i] = table[t * HEADS + h];
    }
}

// ---------------------------------------------------------------------------
// QKV GEMM:  (49152 x 256) x (768 x 256)^T  + bias
// Block = 256 threads (8 waves). Block tile: 256 rows x 64 out-channels.
// Wave tile: 32 rows x 64 cols -> 2 A frags + 4 shared B frags per K-step,
// 8 WMMA per K-step (1 ds_load_b128 per WMMA). Weights staged in LDS.
// Epilogue scatters Q (scaled, bf16), K (bf16), V^T (bf16).
// ---------------------------------------------------------------------------
__global__ __launch_bounds__(256) void qkv_gemm_kernel(
    const bf16* __restrict__ xbf,      // (49152, 256)
    const bf16* __restrict__ wbf,      // (768, 256) = qkv_w in bf16
    const float* __restrict__ qkv_b,   // (768,)
    bf16* __restrict__ Qs,             // (192,8,256,32)  pre-scaled
    bf16* __restrict__ Kf,             // (192,8,256,32)
    bf16* __restrict__ Vt)             // (192,8,32,256)
{
    __shared__ bf16 Wsh[64 * 256];

    const int oc0   = blockIdx.y * 64;           // 0..704
    const int row0b = blockIdx.x * 256;

    // stage 64 weight rows (32 KB) into LDS
    {
        const uint4* src = (const uint4*)(wbf + oc0 * 256);
        uint4* dst = (uint4*)Wsh;
#pragma unroll
        for (int i = threadIdx.x; i < 2048; i += 256) dst[i] = src[i];
    }
    __syncthreads();

    const int lane = threadIdx.x & 31;
    const int wv   = threadIdx.x >> 5;
    const int half = lane >> 4;
    const int lm   = lane & 15;
    const int row0 = row0b + wv * 32;            // wave owns 32 rows

    v8f acc[2][4];
#pragma unroll
    for (int g = 0; g < 2; ++g)
#pragma unroll
        for (int c = 0; c < 4; ++c) acc[g][c] = vzero8();

#pragma unroll
    for (int kt = 0; kt < 8; ++kt) {
        const int k0 = kt * 32;
        v16bf a0 = load_frag_a(xbf + (size_t)(row0 + lm) * 256 + k0, half);
        v16bf a1 = load_frag_a(xbf + (size_t)(row0 + 16 + lm) * 256 + k0, half);
        v16bf bfr[4];
#pragma unroll
        for (int c = 0; c < 4; ++c)
            bfr[c] = load_frag_b(Wsh + (c * 16 + lm) * 256 + k0, half);
#pragma unroll
        for (int c = 0; c < 4; ++c) acc[0][c] = wmma_bf(a0, bfr[c], acc[0][c]);
#pragma unroll
        for (int c = 0; c < 4; ++c) acc[1][c] = wmma_bf(a1, bfr[c], acc[1][c]);
    }

    const float scale = 0.17677669529663687f;    // 32^-0.5
#pragma unroll
    for (int c = 0; c < 4; ++c) {
        const int oc  = oc0 + c * 16 + lm;       // output channel for this lane
        const float bv = qkv_b[oc];
        const int sec = oc >> 8;                 // 0=Q 1=K 2=V (uniform per c)
        const int ocl = oc & 255;
        const int h = ocl >> 5, d = ocl & 31;
#pragma unroll
        for (int g = 0; g < 2; ++g) {
            const int rowg = row0 + g * 16;
#pragma unroll
            for (int r = 0; r < 8; ++r) {
                const int gg = rowg + r + half * 8;   // global token row
                const int b = gg >> 8, n = gg & 255;
                float v = acc[g][c][r] + bv;
                if (sec == 0)
                    Qs[(((size_t)(b * HEADS + h) * NTOK + n) * HD) + d] = f2bf(v * scale);
                else if (sec == 1)
                    Kf[(((size_t)(b * HEADS + h) * NTOK + n) * HD) + d] = f2bf(v);
                else
                    Vt[(((size_t)(b * HEADS + h) * HD + d) * NTOK) + n] = f2bf(v);
            }
        }
    }
}

// ---------------------------------------------------------------------------
// Attention kernel: one block per (b,h). 512 threads = 16 waves, wave w owns
// query rows [16w, 16w+16). S = Q K^T (1 WMMA per 16-col tile, K=hd=32),
// +bias +mask in f32, full-row softmax (16-lane shfl_xor reductions),
// P (bf16) -> LDS -> P V via WMMA (K=256).
// Dynamic LDS: K 16KB | V^T 16KB | P 128KB = 160KB.
// ---------------------------------------------------------------------------
__global__ __launch_bounds__(512) void attn_kernel(
    const bf16* __restrict__ Qs, const bf16* __restrict__ Kf,
    const bf16* __restrict__ Vt,
    const float* __restrict__ biasH,   // (8,256,256)
    const float* __restrict__ maskp,   // (64,256,256)
    bf16* __restrict__ ctx)            // (192,256,256)
{
    extern __shared__ char smem[];
    bf16* Ksh = (bf16*)smem;                   // 256 x 32
    bf16* Vsh = (bf16*)(smem + 16384);         // 32 x 256
    bf16* Psh = (bf16*)(smem + 32768);         // 16 waves x 16 x 256

    const int bh = blockIdx.x;                 // b*8 + h
    const int b  = bh >> 3, h = bh & 7;
    const int wmsk = b & 63;                   // window mask index (b % 64)

    // stage K and V^T tiles (16 KB each)
    {
        const uint4* s1 = (const uint4*)(Kf + (size_t)bh * NTOK * HD);
        const uint4* s2 = (const uint4*)(Vt + (size_t)bh * HD * NTOK);
        uint4* d1 = (uint4*)Ksh;
        uint4* d2 = (uint4*)Vsh;
#pragma unroll
        for (int i = threadIdx.x; i < 1024; i += 512) { d1[i] = s1[i]; d2[i] = s2[i]; }
    }
    __syncthreads();

    const int lane = threadIdx.x & 31;
    const int wv   = threadIdx.x >> 5;         // 0..15
    const int half = lane >> 4;
    const int lm   = lane & 15;
    const int n0   = wv * 16;

    // Q fragment for this wave's 16 rows (K = hd = 32 exactly one WMMA K-step)
    v16bf aq = load_frag_a(Qs + (size_t)bh * NTOK * HD + (n0 + lm) * HD, half);

    // ---- S = Q K^T  (batch B-frag loads in groups of 4 to overlap DS/WMMA) --
    v8f s[16];
#pragma unroll
    for (int tb = 0; tb < 4; ++tb) {
        v16bf bk[4];
#pragma unroll
        for (int j = 0; j < 4; ++j)
            bk[j] = load_frag_b(Ksh + ((tb * 4 + j) * 16 + lm) * HD, half);
#pragma unroll
        for (int j = 0; j < 4; ++j)
            s[tb * 4 + j] = wmma_bf(aq, bk[j], vzero8());
    }

    // ---- + rel-pos bias + shift mask (f32, full precision) ----
    const float* bp = biasH + (size_t)h * NTOK * NTOK;
    const float* mp = maskp + (size_t)wmsk * NTOK * NTOK;
#pragma unroll
    for (int t = 0; t < 16; ++t) {
        const int m = t * 16 + lm;
#pragma unroll
        for (int r = 0; r < 8; ++r) {
            const int n = n0 + r + half * 8;
            s[t][r] += bp[n * NTOK + m] + mp[n * NTOK + m];
        }
    }

    // ---- softmax per row (row lives on a 16-lane group) ----
    bf16* Pw = Psh + wv * 16 * 256;
#pragma unroll
    for (int r = 0; r < 8; ++r) {
        float mx = s[0][r];
#pragma unroll
        for (int t = 1; t < 16; ++t) mx = fmaxf(mx, s[t][r]);
#pragma unroll
        for (int i = 1; i < 16; i <<= 1) mx = fmaxf(mx, __shfl_xor(mx, i, 32));
        float ev[16];
        float sum = 0.0f;
#pragma unroll
        for (int t = 0; t < 16; ++t) { ev[t] = __expf(s[t][r] - mx); sum += ev[t]; }
#pragma unroll
        for (int i = 1; i < 16; i <<= 1) sum += __shfl_xor(sum, i, 32);
        const float inv = 1.0f / sum;
        const int nl = r + half * 8;           // local row 0..15
#pragma unroll
        for (int t = 0; t < 16; ++t)
            Pw[nl * 256 + t * 16 + lm] = f2bf(ev[t] * inv);
    }
    __syncthreads();

    // ---- O = P V  (K = 256, 8 WMMA steps, 2 column tiles of hd) ----
    v8f o0 = vzero8(), o1 = vzero8();
#pragma unroll
    for (int kt = 0; kt < 8; ++kt) {
        const int k0 = kt * 32;
        v16bf ap = load_frag_a(Pw + lm * 256 + k0, half);
        v16bf b0 = load_frag_b(Vsh + lm * 256 + k0, half);          // d = lm
        v16bf b1 = load_frag_b(Vsh + (16 + lm) * 256 + k0, half);   // d = 16+lm
        o0 = wmma_bf(ap, b0, o0);
        o1 = wmma_bf(ap, b1, o1);
    }

    // ---- store context (B_, N, C) bf16 with channel = h*32 + d ----
    bf16* cb = ctx + (size_t)b * NTOK * CDIM + h * HD;
#pragma unroll
    for (int r = 0; r < 8; ++r) {
        const int n = n0 + r + half * 8;
        cb[(size_t)n * CDIM + lm]      = f2bf(o0[r]);
        cb[(size_t)n * CDIM + 16 + lm] = f2bf(o1[r]);
    }
}

// ---------------------------------------------------------------------------
// Output projection GEMM: (49152 x 256) x (256 x 256)^T + bias -> f32 out
// Same structure as QKV GEMM (32x64 wave tile).
// ---------------------------------------------------------------------------
__global__ __launch_bounds__(256) void proj_gemm_kernel(
    const bf16* __restrict__ ctx,      // (49152, 256)
    const bf16* __restrict__ pwbf,     // (256, 256)
    const float* __restrict__ proj_b,  // (256,)
    float* __restrict__ out)           // (49152, 256) f32
{
    __shared__ bf16 Wsh[64 * 256];

    const int oc0   = blockIdx.y * 64;
    const int row0b = blockIdx.x * 256;

    {
        const uint4* src = (const uint4*)(pwbf + oc0 * 256);
        uint4* dst = (uint4*)Wsh;
#pragma unroll
        for (int i = threadIdx.x; i < 2048; i += 256) dst[i] = src[i];
    }
    __syncthreads();

    const int lane = threadIdx.x & 31;
    const int wv   = threadIdx.x >> 5;
    const int half = lane >> 4;
    const int lm   = lane & 15;
    const int row0 = row0b + wv * 32;

    v8f acc[2][4];
#pragma unroll
    for (int g = 0; g < 2; ++g)
#pragma unroll
        for (int c = 0; c < 4; ++c) acc[g][c] = vzero8();

#pragma unroll
    for (int kt = 0; kt < 8; ++kt) {
        const int k0 = kt * 32;
        v16bf a0 = load_frag_a(ctx + (size_t)(row0 + lm) * 256 + k0, half);
        v16bf a1 = load_frag_a(ctx + (size_t)(row0 + 16 + lm) * 256 + k0, half);
        v16bf bfr[4];
#pragma unroll
        for (int c = 0; c < 4; ++c)
            bfr[c] = load_frag_b(Wsh + (c * 16 + lm) * 256 + k0, half);
#pragma unroll
        for (int c = 0; c < 4; ++c) acc[0][c] = wmma_bf(a0, bfr[c], acc[0][c]);
#pragma unroll
        for (int c = 0; c < 4; ++c) acc[1][c] = wmma_bf(a1, bfr[c], acc[1][c]);
    }

#pragma unroll
    for (int c = 0; c < 4; ++c) {
        const int oc = oc0 + c * 16 + lm;
        const float pb = proj_b[oc];
#pragma unroll
        for (int g = 0; g < 2; ++g) {
            const int rowg = row0 + g * 16;
#pragma unroll
            for (int r = 0; r < 8; ++r) {
                const int gg = rowg + r + half * 8;
                out[(size_t)gg * CDIM + oc] = acc[g][c][r] + pb;
            }
        }
    }
}

// ---------------------------------------------------------------------------
// Host launcher
// ---------------------------------------------------------------------------
extern "C" void kernel_launch(void* const* d_in, const int* in_sizes, int n_in,
                              void* d_out, int out_size, void* d_ws, size_t ws_size,
                              hipStream_t stream) {
    const float* x        = (const float*)d_in[0];   // (192,256,256)
    const float* mask     = (const float*)d_in[1];   // (64,256,256)
    const float* qkv_w    = (const float*)d_in[2];   // (768,256)
    const float* qkv_b    = (const float*)d_in[3];   // (768,)
    const float* proj_w   = (const float*)d_in[4];   // (256,256)
    const float* proj_b   = (const float*)d_in[5];   // (256,)
    const float* rp_table = (const float*)d_in[6];   // (1575,8)
    const int*   rp_index = (const int*)d_in[7];     // (256,256)
    float* out = (float*)d_out;

    char* ws = (char*)d_ws;
    const size_t SZ_TOK_BF = (size_t)MTOT * CDIM * sizeof(bf16);   // 25,165,824
    bf16*  xbf   = (bf16*)(ws);
    bf16*  Qs    = (bf16*)(ws + SZ_TOK_BF);
    bf16*  Kf    = (bf16*)(ws + 2 * SZ_TOK_BF);
    bf16*  Vt    = (bf16*)(ws + 3 * SZ_TOK_BF);
    bf16*  ctx   = (bf16*)(ws + 4 * SZ_TOK_BF);
    bf16*  wbf   = (bf16*)(ws + 5 * SZ_TOK_BF);                    // 768*256 bf16
    bf16*  pwbf  = (bf16*)(ws + 5 * SZ_TOK_BF + 768 * 256 * sizeof(bf16));
    float* biasH = (float*)(ws + 5 * SZ_TOK_BF + (768 + 256) * 256 * sizeof(bf16));

    // 1) precision conversion + bias gather
    {
        int n = MTOT * CDIM;
        f32_to_bf16_kernel<<<(n + 255) / 256, 256, 0, stream>>>(x, xbf, n);
        n = 768 * 256;
        f32_to_bf16_kernel<<<(n + 255) / 256, 256, 0, stream>>>(qkv_w, wbf, n);
        n = 256 * 256;
        f32_to_bf16_kernel<<<(n + 255) / 256, 256, 0, stream>>>(proj_w, pwbf, n);
        bias_gather_kernel<<<(NTOK * NTOK + 255) / 256, 256, 0, stream>>>(
            rp_table, rp_index, biasH);
    }

    // 2) QKV projection (rows 49152, OC 768): 192 x 12 blocks of 256 threads
    {
        dim3 grid(MTOT / 256, 768 / 64);
        qkv_gemm_kernel<<<grid, 256, 0, stream>>>(xbf, wbf, qkv_b, Qs, Kf, Vt);
    }

    // 3) attention: one block per (b,h); 160 KB dynamic LDS
    attn_kernel<<<NWIN * HEADS, 512, 163840, stream>>>(Qs, Kf, Vt, biasH, mask, ctx);

    // 4) output projection: 192 x 4 blocks
    {
        dim3 grid(MTOT / 256, CDIM / 64);
        proj_gemm_kernel<<<grid, 256, 0, stream>>>(ctx, pwbf, proj_b, out);
    }
    (void)in_sizes; (void)n_in; (void)out_size; (void)ws_size;
}
